// Correlation_70884140253505
// MI455X (gfx1250) — compile-verified, matrix operand order checked by reading, not run on hardware
//
#include <hip/hip_runtime.h>

// FlowNetC correlation, fixed shapes from reference:
//   in1, in2: (8, 256, 64, 128) f32 ; out: (8, 81, 64, 128) f32
#define B_  8
#define C_  256
#define H_  64
#define W_  128
#define HW_ (H_ * W_)

typedef __attribute__((ext_vector_type(16))) _Float16 v16h;
typedef __attribute__((ext_vector_type(8)))  float    v8f;

union Frag16 { v16h v; unsigned u[8]; };
union Acc8   { v8f  v; float    f[8]; };

__device__ __forceinline__ unsigned short f2h_bits(float f) {
    _Float16 h = (_Float16)f;
    union { _Float16 h; unsigned short s; } r;
    r.h = h;
    return r.s;
}

__global__ __launch_bounds__(256, 1)
void corr_wmma_f16_kernel(const float* __restrict__ in1,
                          const float* __restrict__ in2,
                          float* __restrict__ out) {
    // LDS staging, x-major so K is contiguous -> packed b32 fragment loads.
    // Row stride 68 ushorts (136 B): 8B-aligned rows, odd 8B stride (34 dwords) ->
    // conflict-free strided fragment reads.
    __shared__ unsigned short aT[W_][68];        // in1 chunk: [x][c], c in [0,64)
    __shared__ unsigned short bT[W_ + 16][68];   // padded in2 row chunk: [p][c], p in [0,144)
    __shared__ float sc[8][16][34];              // per-wave 16x32 (+pad) output tiles

    const int tid  = threadIdx.x;
    const int lane = tid & 31;
    const int wave = tid >> 5;
    const int hi   = lane >> 4;        // 0: lanes 0-15, 1: lanes 16-31
    const int mn   = lane & 15;        // M for A-frag, N for B-frag
    const int koff = hi << 3;          // hi lanes hold K+8 within each 16-K half
    const int x0   = wave << 4;        // this wave's 16-pixel tile

    const int bb = blockIdx.x / H_;
    const int y  = blockIdx.x - bb * H_;

    const size_t base1 = (size_t)bb * C_ * HW_ + (size_t)y * W_;

    // One-time zero fill of bT's 16 pad columns (p in {0..3, 132..143}); the
    // interior staging below only ever writes p in [4,132), so these stay zero.
    for (int i = tid; i < 512; i += 256) {
        const int pi = i >> 5;                 // 0..15
        const int cq = (i & 31) * 2;           // even channel index
        const int p  = (pi < 4) ? pi : (128 + pi);
        *(unsigned*)&bT[p][cq] = 0u;
    }

    // 18 live accumulator tiles: 9 dy x (2 N-tiles of 16x16 f32) = 144 VGPRs.
    Acc8 acc0[9], acc1[9];
    #pragma unroll
    for (int dy = 0; dy < 9; ++dy)
        #pragma unroll
        for (int v = 0; v < 8; ++v) { acc0[dy].f[v] = 0.0f; acc1[dy].f[v] = 0.0f; }

    for (int kc = 0; kc < C_; kc += 64) {
        __syncthreads();   // previous chunk's aT/bT reads done everywhere

        // Stage A chunk once per kc: in1[b, kc+c, y, x..x+3] -> aT[x..x+3][c].
        for (int i = tid; i < 2048; i += 256) {
            const int c = i >> 5;
            const int x = (i & 31) << 2;
            const size_t ga = base1 + (size_t)(kc + c) * HW_ + x;
            const float4 f = *(const float4*)&in1[ga];
            if (kc + 64 < C_)
                __builtin_prefetch(&in1[ga + (size_t)64 * HW_], 0, 0);
            aT[x + 0][c] = f2h_bits(f.x);
            aT[x + 1][c] = f2h_bits(f.y);
            aT[x + 2][c] = f2h_bits(f.z);
            aT[x + 3][c] = f2h_bits(f.w);
        }

        #pragma unroll
        for (int dy = 0; dy < 9; ++dy) {
            const int  row   = y + dy - 4;
            const bool rowok = (row >= 0) && (row < H_);
            const size_t base2 = (size_t)bb * C_ * HW_ + (size_t)row * W_; // used iff rowok

            __syncthreads();   // previous dy's bT reads done (also orders aT writes)

            // Stage B chunk: interior p = x+4, x multiple of 4, vectorized.
            for (int i = tid; i < 2048; i += 256) {
                const int c = i >> 5;
                const int x = (i & 31) << 2;
                float4 f = make_float4(0.0f, 0.0f, 0.0f, 0.0f);
                if (rowok)
                    f = *(const float4*)&in2[base2 + (size_t)(kc + c) * HW_ + x];
                const int p = x + 4;
                bT[p + 0][c] = f2h_bits(f.x);
                bT[p + 1][c] = f2h_bits(f.y);
                bT[p + 2][c] = f2h_bits(f.z);
                bT[p + 3][c] = f2h_bits(f.w);
            }
            __syncthreads();

            // Two K=32 WMMA steps per 64-channel chunk, accumulate tiles for this dy.
            #pragma unroll
            for (int ks = 0; ks < 2; ++ks) {
                Frag16 a, b0, b1;
                #pragma unroll
                for (int v = 0; v < 8; ++v) {
                    // ISA 16-bit fragment layout: VGPR v<4 -> K=2v, v>=4 -> K=2v+8;
                    // hi-lanes shifted by +8. Pair (K, K+1) packed in one dword.
                    const int kb = (((v < 4) ? (2 * v) : (2 * v + 8)) + koff) + (ks << 5);
                    a.u[v]  = *(const unsigned*)&aT[x0 + mn][kb];
                    b0.u[v] = *(const unsigned*)&bT[x0 + mn][kb];
                    b1.u[v] = *(const unsigned*)&bT[x0 + 16 + mn][kb];
                }
                acc0[dy].v = __builtin_amdgcn_wmma_f32_16x16x32_f16(
                    false, a.v, false, b0.v, (short)0, acc0[dy].v, false, false);
                acc1[dy].v = __builtin_amdgcn_wmma_f32_16x16x32_f16(
                    false, a.v, false, b1.v, (short)0, acc1[dy].v, false, false);
            }
        }
    }

    // Epilogue: per dy, spill the two 16x16 tiles to LDS (C/D layout: lane n,
    // VGPR v -> M = v + 8*hi, N = n), then extract the banded diagonal:
    //   out[x0+m, dx] = tile[m][m + j4], j4 = dx+4 in [0,9).
    #pragma unroll
    for (int dy = 0; dy < 9; ++dy) {
        #pragma unroll
        for (int v = 0; v < 8; ++v) {
            const int m = v + (hi << 3);
            sc[wave][m][mn]      = acc0[dy].f[v];
            sc[wave][m][16 + mn] = acc1[dy].f[v];
        }
        for (int e = lane; e < 144; e += 32) {
            const int j4 = e >> 4;     // 0..8
            const int m  = e & 15;     // coalesced over 16 consecutive x
            const float val = sc[wave][m][m + j4] * (1.0f / (float)C_);
            const int d = dy * 9 + j4;
            out[(((size_t)bb * 81 + d) * H_ + y) * W_ + (x0 + m)] = val;
        }
    }
}

extern "C" void kernel_launch(void* const* d_in, const int* in_sizes, int n_in,
                              void* d_out, int out_size, void* d_ws, size_t ws_size,
                              hipStream_t stream) {
    (void)in_sizes; (void)n_in; (void)out_size; (void)d_ws; (void)ws_size;
    const float* in1 = (const float*)d_in[0];
    const float* in2 = (const float*)d_in[1];
    float* out = (float*)d_out;
    corr_wmma_f16_kernel<<<dim3(B_ * H_), dim3(256), 0, stream>>>(in1, in2, out);
}
